// CausalSelfAttention_27444841022079
// MI455X (gfx1250) — compile-verified, matrix operand order checked by reading, not run on hardware
//
#include <hip/hip_runtime.h>
#include <math.h>

typedef __attribute__((ext_vector_type(16))) __bf16 v16bf;
typedef __attribute__((ext_vector_type(8)))  __bf16 v8bf;   // 16 bytes
typedef __attribute__((ext_vector_type(4)))  __bf16 v4bf;   // 8 bytes
typedef __attribute__((ext_vector_type(8)))  float  v8f;

#define DIMC   2048
#define NHEADS 16
#define HSIZE  128
#define TSEQ   2048
#define BATCH  2
#define STR    136   // LDS row stride (bf16 elems) for 128-wide tiles; 272B = 17*16B
#define ASTR   40    // LDS row stride for 32-wide tiles; 80B = 5*16B

// ---------------- CDNA5 async global->LDS copy (ASYNCcnt path, ISA 15.18.3 op 98) ----------------
// Per-lane 16B DMA into LDS, no VGPR data movement. LDS operand is the byte offset
// within the LDS aperture (flat shared address truncated to 32 bits, ISA 10.2).
__device__ __forceinline__ void async_copy_b128(void* lds_dst, const void* gsrc) {
  unsigned lds_off = (unsigned)(uintptr_t)lds_dst;
  asm volatile("global_load_async_to_lds_b128 %0, %1, off"
               :: "v"(lds_off), "v"(gsrc)
               : "memory");
}
__device__ __forceinline__ void wait_async0() {
  asm volatile("s_wait_asynccnt 0x0" ::: "memory");
}

// ---------------- WMMA fragment loader (ISA 7.12.2 16-bit A layout, wave32) ----------------
// Works for: A row-major [m][k], B stored transposed [n][k], K^T from row-major K [kv][d].
// Lane holds row m=lane&15; lanes16-31 offset K by +8; VGPR0-3 = K..K+7, VGPR4-7 = K+16..K+23.
// Each lane: two contiguous 16-byte LDS loads -> 2x ds_load_b128.
__device__ __forceinline__ v16bf frag_ld16x32(const __bf16* base, int stride, int m0, int k0, int lane) {
  const int m  = m0 + (lane & 15);
  const int kb = k0 + ((lane >> 4) << 3);
  const __bf16* p = base + m * stride + kb;
  const v8bf lo = *(const v8bf*)p;
  const v8bf hi = *(const v8bf*)(p + 16);
  return __builtin_shufflevector(lo, hi, 0,1,2,3,4,5,6,7,8,9,10,11,12,13,14,15);
}

__device__ __forceinline__ v8f v8f_zero() {
  v8f z = {0.f,0.f,0.f,0.f,0.f,0.f,0.f,0.f};
  return z;
}

// ---------------- GEMM: C[MxN] = A[MxK] @ B[KxN], bf16 WMMA, f32 accum ----------------
// Block: 256 threads, tile 128x128, K-step 32. 8 waves in 2x4; wave tile 64x32.
// sA row-major [m][k]; sB stored TRANSPOSED [n][k] so B fragments are contiguous loads.
// (Fills must pass through VGPRs anyway for the f32 -> bf16 conversion, so no async here.)
template<typename TA, typename TB, bool OUT_BF16>
__global__ __launch_bounds__(256) void gemm_wmma_kernel(const TA* __restrict__ A,
                                                        const TB* __restrict__ B,
                                                        void* __restrict__ Cv,
                                                        int M, int N, int K) {
  __shared__ __bf16 sA [128 * ASTR];   // 128 m x 32 k
  __shared__ __bf16 sBt[128 * ASTR];   // 128 n x 32 k (transposed)

  const int tid  = threadIdx.x;
  const int lane = tid & 31;
  const int wid  = tid >> 5;
  const int wm   = wid >> 2;     // 0..1
  const int wn   = wid & 3;      // 0..3
  const int m0   = blockIdx.y * 128;
  const int n0   = blockIdx.x * 128;

  v8f acc[4][2];
#pragma unroll
  for (int i = 0; i < 4; ++i)
#pragma unroll
    for (int j = 0; j < 2; ++j) acc[i][j] = v8f_zero();

  for (int k0 = 0; k0 < K; k0 += 32) {
    __syncthreads();
    // A fill: 4 consecutive k per thread -> one 8B LDS store each
    for (int i = tid * 4; i < 128 * 32; i += 256 * 4) {
      const int r = i >> 5, c = i & 31;
      v4bf t;
#pragma unroll
      for (int e = 0; e < 4; ++e) t[e] = (__bf16)(float)A[(size_t)(m0 + r) * K + (k0 + c + e)];
      *(v4bf*)(&sA[r * ASTR + c]) = t;
    }
    // B fill (transposed): 4 consecutive n per thread, coalesced global reads
    for (int i = tid * 4; i < 32 * 128; i += 256 * 4) {
      const int r = i >> 7, c = i & 127;   // r = k, c = n (multiple of 4)
#pragma unroll
      for (int e = 0; e < 4; ++e)
        sBt[(c + e) * ASTR + r] = (__bf16)(float)B[(size_t)(k0 + r) * N + (n0 + c + e)];
    }
    // prefetch next K-step while this tile is consumed
    if (k0 + 32 < K) {
      __builtin_prefetch(&A[(size_t)(m0 + (tid & 127)) * K + (k0 + 32)], 0, 3);
      __builtin_prefetch(&B[(size_t)(k0 + 32 + (tid & 31)) * N + n0 + ((tid >> 5) << 4)], 0, 3);
    }
    __syncthreads();

    v16bf af[4];
#pragma unroll
    for (int i = 0; i < 4; ++i) af[i] = frag_ld16x32(sA, ASTR, wm * 64 + i * 16, 0, lane);
    v16bf bfr[2];
#pragma unroll
    for (int j = 0; j < 2; ++j) bfr[j] = frag_ld16x32(sBt, ASTR, wn * 32 + j * 16, 0, lane);

#pragma unroll
    for (int i = 0; i < 4; ++i)
#pragma unroll
      for (int j = 0; j < 2; ++j)
        acc[i][j] = __builtin_amdgcn_wmma_f32_16x16x32_bf16(
            false, af[i], false, bfr[j], (short)0, acc[i][j], false, false);
  }

  const int hi = lane >> 4;
  const int ln = lane & 15;
#pragma unroll
  for (int i = 0; i < 4; ++i)
#pragma unroll
    for (int j = 0; j < 2; ++j)
#pragma unroll
      for (int r = 0; r < 8; ++r) {
        const size_t row = (size_t)(m0 + wm * 64 + i * 16 + r + 8 * hi);
        const size_t col = (size_t)(n0 + wn * 32 + j * 16 + ln);
        if constexpr (OUT_BF16) ((__bf16*)Cv)[row * N + col] = (__bf16)acc[i][j][r];
        else                    ((float*)Cv)[row * N + col]  = acc[i][j][r];
      }
}

// ---------------- RoPE on q,k inside qkv (bf16), folds softmax scale into q ----------------
__global__ __launch_bounds__(256) void rope_kernel(__bf16* __restrict__ qkv,
                                                   const float* __restrict__ freqs,
                                                   float scale) {
  const int idx = blockIdx.x * 256 + threadIdx.x;  // over B*T*NHEADS*(HSIZE/2)
  const int p = idx & 63;
  const int h = (idx >> 6) & 15;
  const int t = (idx >> 10) & 2047;
  const int b = idx >> 21;
  if (b >= BATCH) return;

  const float c = freqs[(t * 64 + p) * 2 + 0];
  const float s = freqs[(t * 64 + p) * 2 + 1];
  const size_t rowbase = (size_t)(b * TSEQ + t) * (3 * DIMC);
  const size_t qoff = rowbase + (size_t)h * HSIZE + 2 * p;
  const size_t koff = qoff + DIMC;

  const float q0 = (float)qkv[qoff], q1 = (float)qkv[qoff + 1];
  const float k0 = (float)qkv[koff], k1 = (float)qkv[koff + 1];
  qkv[qoff]     = (__bf16)((q0 * c - q1 * s) * scale);
  qkv[qoff + 1] = (__bf16)((q0 * s + q1 * c) * scale);
  qkv[koff]     = (__bf16)(k0 * c - k1 * s);
  qkv[koff + 1] = (__bf16)(k0 * s + k1 * c);
}

// ---------------- Flash attention: one block per (b, h, 128-row q tile) ----------------
// 256 threads = 8 waves; wave w owns q rows [w*16, w*16+16). Online softmax, WMMA for
// S = Q K^T and O += P V. sQ/sK row-major (staged via async LDS DMA); sV stored
// TRANSPOSED [d][kv] (needs the register pass); sP row-major.
__global__ __launch_bounds__(256) void attn_kernel(const __bf16* __restrict__ qkv,
                                                   __bf16* __restrict__ y) {
  extern __shared__ char smem[];
  __bf16* sQ  = (__bf16*)smem;
  __bf16* sK  = sQ  + 128 * STR;   // [kv][d] — used as K^T via transposed fragment indexing
  __bf16* sVt = sK  + 128 * STR;   // [d][kv]
  __bf16* sP  = sVt + 128 * STR;   // [qrow][kv]

  const int tid  = threadIdx.x;
  const int lane = tid & 31;
  const int wid  = tid >> 5;
  const int hi   = lane >> 4;
  const int ln   = lane & 15;

  const int qb = blockIdx.x;
  const int h  = blockIdx.y;
  const int b  = blockIdx.z;

  const size_t rowstride = 3 * DIMC;
  const size_t qbase = ((size_t)(b * TSEQ) + (size_t)qb * 128) * rowstride + (size_t)h * HSIZE;

  // stage Q tile (128 x 128 bf16) via async global->LDS DMA (no VGPR round-trip)
  for (int i = tid; i < 128 * 16; i += 256) {
    const int r = i >> 4, c4 = i & 15;
    async_copy_b128(sQ + r * STR + c4 * 8, qkv + qbase + (size_t)r * rowstride + c4 * 8);
  }

  v8f o[8];
#pragma unroll
  for (int dt = 0; dt < 8; ++dt) o[dt] = v8f_zero();
  float rmax[8], rsum[8];
#pragma unroll
  for (int j = 0; j < 8; ++j) { rmax[j] = -3.0e38f; rsum[j] = 0.0f; }

  const int qrow0 = qb * 128 + wid * 16;

  for (int kvb = 0; kvb <= qb; ++kvb) {
    __syncthreads();   // prior iter done with sK/sVt/sP
    const size_t kbase = ((size_t)(b * TSEQ) + (size_t)kvb * 128) * rowstride + DIMC + (size_t)h * HSIZE;
    const size_t vbase = kbase + DIMC;
    // K tile: verbatim copy -> async DMA. V tile: must transpose -> through VGPRs.
    for (int i = tid; i < 128 * 16; i += 256) {
      const int r = i >> 4, c4 = i & 15;   // r = kv row, c4*8 = d col base
      async_copy_b128(sK + r * STR + c4 * 8, qkv + kbase + (size_t)r * rowstride + c4 * 8);
      const uint4 vv = *(const uint4*)(qkv + vbase + (size_t)r * rowstride + c4 * 8);
      const __bf16* ve = (const __bf16*)&vv;
#pragma unroll
      for (int e = 0; e < 8; ++e) sVt[(c4 * 8 + e) * STR + r] = ve[e];  // transpose into LDS
    }
    // prefetch next kv block while DMA + transpose are in flight
    if (kvb + 1 <= qb) {
      const size_t nb = kbase + 128 * rowstride;
      __builtin_prefetch(qkv + nb + (size_t)(tid & 127) * rowstride, 0, 3);
      __builtin_prefetch(qkv + nb + DIMC + (size_t)(tid & 127) * rowstride, 0, 3);
    }
    wait_async0();      // this wave's async copies (Q on first iter, K always) landed
    __syncthreads();    // all waves' copies + V transpose visible

    // S = Q K^T for this wave's 16 rows x 128 cols (scale already folded into q)
    v16bf aq[4];
#pragma unroll
    for (int dk = 0; dk < 4; ++dk) aq[dk] = frag_ld16x32(sQ, STR, wid * 16, dk * 32, lane);

    v8f s[8];
#pragma unroll
    for (int nt = 0; nt < 8; ++nt) {
      s[nt] = v8f_zero();
#pragma unroll
      for (int dk = 0; dk < 4; ++dk)
        s[nt] = __builtin_amdgcn_wmma_f32_16x16x32_bf16(
            false, aq[dk], false, frag_ld16x32(sK, STR, nt * 16, dk * 32, lane),
            (short)0, s[nt], false, false);
    }

    // causal mask (only the diagonal block can be partially masked)
    if (kvb == qb) {
#pragma unroll
      for (int nt = 0; nt < 8; ++nt)
#pragma unroll
        for (int j = 0; j < 8; ++j) {
          const int col = kvb * 128 + nt * 16 + ln;
          const int row = qrow0 + j + 8 * hi;
          if (col > row) s[nt][j] = -3.0e38f;
        }
    }

    // online softmax: row spans 16 lanes of one half-wave at fixed VGPR slot j
    float bmax[8];
#pragma unroll
    for (int j = 0; j < 8; ++j) {
      bmax[j] = s[0][j];
#pragma unroll
      for (int nt = 1; nt < 8; ++nt) bmax[j] = fmaxf(bmax[j], s[nt][j]);
    }
#pragma unroll
    for (int m = 1; m <= 8; m <<= 1)
#pragma unroll
      for (int j = 0; j < 8; ++j) bmax[j] = fmaxf(bmax[j], __shfl_xor(bmax[j], m, 32));

    float corr[8];
#pragma unroll
    for (int j = 0; j < 8; ++j) {
      const float nm = fmaxf(rmax[j], bmax[j]);
      corr[j] = __expf(rmax[j] - nm);
      rmax[j] = nm;
    }

    float bsum[8];
#pragma unroll
    for (int j = 0; j < 8; ++j) bsum[j] = 0.0f;
#pragma unroll
    for (int nt = 0; nt < 8; ++nt)
#pragma unroll
      for (int j = 0; j < 8; ++j) {
        const float p = __expf(s[nt][j] - rmax[j]);
        bsum[j] += p;
        sP[(wid * 16 + j + 8 * hi) * STR + nt * 16 + ln] = (__bf16)p;
      }
#pragma unroll
    for (int m = 1; m <= 8; m <<= 1)
#pragma unroll
      for (int j = 0; j < 8; ++j) bsum[j] += __shfl_xor(bsum[j], m, 32);
#pragma unroll
    for (int j = 0; j < 8; ++j) rsum[j] = rsum[j] * corr[j] + bsum[j];

    // rescale running output
#pragma unroll
    for (int dt = 0; dt < 8; ++dt)
#pragma unroll
      for (int j = 0; j < 8; ++j) o[dt][j] *= corr[j];

    __syncthreads();   // sP writes visible (and sVt/sK safe across waves)

    // O += P @ V   (V fragments contiguous thanks to transposed sVt)
    v16bf ap[4];
#pragma unroll
    for (int kk = 0; kk < 4; ++kk) ap[kk] = frag_ld16x32(sP, STR, wid * 16, kk * 32, lane);
#pragma unroll
    for (int dt = 0; dt < 8; ++dt)
#pragma unroll
      for (int kk = 0; kk < 4; ++kk)
        o[dt] = __builtin_amdgcn_wmma_f32_16x16x32_bf16(
            false, ap[kk], false, frag_ld16x32(sVt, STR, dt * 16, kk * 32, lane),
            (short)0, o[dt], false, false);
  }

  // normalize and store y (bf16, row-major B*T x DIM with head offset)
  float inv[8];
#pragma unroll
  for (int j = 0; j < 8; ++j) inv[j] = (rsum[j] > 0.0f) ? (1.0f / rsum[j]) : 0.0f;

  const size_t ybase = ((size_t)(b * TSEQ) + (size_t)qb * 128) * DIMC + (size_t)h * HSIZE;
#pragma unroll
  for (int dt = 0; dt < 8; ++dt)
#pragma unroll
    for (int j = 0; j < 8; ++j) {
      const int row = wid * 16 + j + 8 * hi;
      y[ybase + (size_t)row * DIMC + dt * 16 + ln] = (__bf16)(o[dt][j] * inv[j]);
    }
}

// ---------------- host ----------------
extern "C" void kernel_launch(void* const* d_in, const int* in_sizes, int n_in,
                              void* d_out, int out_size, void* d_ws, size_t ws_size,
                              hipStream_t stream) {
  const float* x      = (const float*)d_in[0];
  const float* freqs  = (const float*)d_in[1];
  const float* w_attn = (const float*)d_in[2];
  const float* w_proj = (const float*)d_in[3];

  char* ws = (char*)d_ws;
  __bf16* qkv = (__bf16*)ws;                                         // (B*T) x 3*DIM bf16
  __bf16* y   = (__bf16*)(ws + (size_t)BATCH * TSEQ * 3 * DIMC * 2); // (B*T) x DIM bf16

  const int M = BATCH * TSEQ;     // 4096

  // 1) qkv = x @ w_attn  (f32 in, bf16 out)
  gemm_wmma_kernel<float, float, true>
      <<<dim3((3 * DIMC) / 128, M / 128), 256, 0, stream>>>(x, w_attn, qkv, M, 3 * DIMC, DIMC);

  // 2) RoPE on q,k (scale folded into q)
  const int total = BATCH * TSEQ * NHEADS * (HSIZE / 2);
  rope_kernel<<<total / 256, 256, 0, stream>>>(qkv, freqs, 1.0f / sqrtf((float)HSIZE));

  // 3) flash attention -> y (bf16)
  const size_t smem = (size_t)4 * 128 * STR * sizeof(__bf16);   // ~136 KB of 320 KB/WGP
  attn_kernel<<<dim3(TSEQ / 128, NHEADS, BATCH), 256, smem, stream>>>(qkv, y);

  // 4) out = y @ w_proj  (bf16 A, f32 B, f32 out)
  gemm_wmma_kernel<__bf16, float, false>
      <<<dim3(DIMC / 128, M / 128), 256, 0, stream>>>(y, w_proj, d_out, M, DIMC, DIMC);
}
